// SlotAttention_10565619548297
// MI455X (gfx1250) — matrix operand order repaired and checked by reading, not compiled, not run on hardware
//
#include <hip/hip_runtime.h>

typedef __attribute__((ext_vector_type(16))) _Float16 v16h;
typedef __attribute__((ext_vector_type(8)))  _Float16 v8h;
typedef __attribute__((ext_vector_type(8)))  float    v8f;

#define B_  32
#define N_  4096
#define D_  256
#define K_  8
#define H_  512
#define BK_ 256
#define LN_EPS 1e-5f
#define EPS_ 1e-8f
#define SCALE_ 0.0625f   // 256^-0.5

// ---------------------------------------------------------------------------
// Generic single-wave WMMA GEMM:  C = act( scale * (A @ Bw^T) + bias )
//   A:  [M, Kdim] f16 row-major (lda), per-batch stride sA (elements)
//   Bw: [N, Kdim] f16 row-major (ldb), per-batch stride sB
//   Each wave (32 threads) computes a 16 x (16*NF) tile.
//   grid = (M/16, N/(16*NF), batches), block = 32.
// OUTMODE: 0 = row-major [ldc] (f32 or f16), batched by sC
//          1 = q-pad remap: row m -> (m/8)*16 + m%8  (f16, ldc)
//          2 = compact slots: batched, keep rows 0..7 only (f16, ldc, sC)
//          3 = transposed store into vT[B][D][N_]: (m,n) -> [m/N_][n][m%N_] (f16)
// ACT: 0 = none, 1 = exact GELU
// ---------------------------------------------------------------------------
template<int NF, int OUTMODE, bool OUT_F16, int ACT>
__global__ __launch_bounds__(32)
void wmma_gemm(const _Float16* __restrict__ A, int lda, size_t sA,
               const _Float16* __restrict__ Bw, int ldb, size_t sB,
               void* __restrict__ C, int ldc, size_t sC,
               int Kdim, const float* __restrict__ bias, float scale)
{
    const int lane = threadIdx.x;
    const int m0 = blockIdx.x * 16;
    const int n0 = blockIdx.y * 16 * NF;
    const int bz = blockIdx.z;

    const _Float16* Ab = A + (size_t)bz * sA;
    const _Float16* Bb = Bw + (size_t)bz * sB;

    // A fragment: lane holds row M = lane%16; element e -> K = 16*(e/8) + 8*(lane/16) + e%8
    const _Float16* arow = Ab + (size_t)(m0 + (lane & 15)) * lda;
    const int half8 = (lane >> 4) * 8;

    // B fragment: lane holds col N = n + lane%16; element e -> K = 16*(e/8) + 8*(lane/16) + e%8
    const _Float16* brow[NF];
#pragma unroll
    for (int j = 0; j < NF; ++j)
        brow[j] = Bb + (size_t)(n0 + j * 16 + (lane & 15)) * ldb + (lane >> 4) * 8;

    v8f acc[NF] = {};

    for (int k0 = 0; k0 < Kdim; k0 += 32) {
        v8h alo = *(const v8h*)(arow + k0 + half8);
        v8h ahi = *(const v8h*)(arow + k0 + 16 + half8);
        v16h a = __builtin_shufflevector(alo, ahi,
                   0,1,2,3,4,5,6,7,8,9,10,11,12,13,14,15);
#pragma unroll
        for (int j = 0; j < NF; ++j) {
            v8h blo = *(const v8h*)(brow[j] + k0);
            v8h bhi = *(const v8h*)(brow[j] + k0 + 16);
            v16h b = __builtin_shufflevector(blo, bhi,
                       0,1,2,3,4,5,6,7,8,9,10,11,12,13,14,15);
            acc[j] = __builtin_amdgcn_wmma_f32_16x16x32_f16(
                         false, a, false, b, (short)0, acc[j], false, false);
        }
    }

    // C/D layout: lane holds col N = n0 + lane%16, rows M = m0 + 8*(lane/16) + r
    const int mbase = m0 + ((lane >> 4) << 3);
#pragma unroll
    for (int j = 0; j < NF; ++j) {
        const int n = n0 + j * 16 + (lane & 15);
        const float bv = bias ? bias[n] : 0.f;
#pragma unroll
        for (int r = 0; r < 8; ++r) {
            float val = acc[j][r] * scale + bv;
            if (ACT == 1) val = 0.5f * val * (1.f + erff(val * 0.70710678f));
            const int m = mbase + r;
            size_t idx;
            if (OUTMODE == 0) {
                idx = (size_t)bz * sC + (size_t)m * ldc + n;
            } else if (OUTMODE == 1) {
                int prow = (m >> 3) * 16 + (m & 7);
                idx = (size_t)prow * ldc + n;
            } else if (OUTMODE == 2) {
                if (lane >= 16) continue;            // rows 8..15 are padding
                idx = (size_t)bz * sC + (size_t)m * ldc + n;
            } else { // 3: vT[B][D][N_]
                idx = (size_t)(m >> 12) * ((size_t)D_ * N_) + (size_t)n * N_ + (m & (N_ - 1));
            }
            if (OUT_F16) ((_Float16*)C)[idx] = (_Float16)val;
            else         ((float*)C)[idx]    = val;
        }
    }
}

// ---------------------------------------------------------------------------
// LayerNorm over 256 columns, fp32 in -> f16 out. grid = rows, block = 256.
// ---------------------------------------------------------------------------
__global__ void ln256(const float* __restrict__ in, const float* __restrict__ g,
                      const float* __restrict__ bta, _Float16* __restrict__ out)
{
    const int row = blockIdx.x, tid = threadIdx.x;
    const float v = in[(size_t)row * 256 + tid];
    __shared__ float s1[256], s2[256];
    s1[tid] = v; s2[tid] = v * v;
    __syncthreads();
    for (int s = 128; s > 0; s >>= 1) {
        if (tid < s) { s1[tid] += s1[tid + s]; s2[tid] += s2[tid + s]; }
        __syncthreads();
    }
    const float mean = s1[0] * (1.f / 256.f);
    const float var  = s2[0] * (1.f / 256.f) - mean * mean;
    const float inv  = rsqrtf(var + LN_EPS);
    out[(size_t)row * 256 + tid] = (_Float16)((v - mean) * inv * g[tid] + bta[tid]);
}

// ---------------------------------------------------------------------------
// Softmax over the 8 slots for each (b, token); +EPS then renormalize.
// Writes fp32 attn to d_out and f16 attn into padded [B][16][N] buffer.
// ---------------------------------------------------------------------------
__global__ void softmax_slots(const float* __restrict__ logits,
                              float* __restrict__ attn_out,
                              _Float16* __restrict__ attn_pad)
{
    const int t = blockIdx.x * blockDim.x + threadIdx.x;
    if (t >= B_ * N_) return;
    const int b = t >> 12, tok = t & (N_ - 1);
    const float* base = logits + (size_t)b * 16 * N_ + tok;
    float l[K_], mx = -1e30f;
#pragma unroll
    for (int r = 0; r < K_; ++r) { l[r] = base[(size_t)r * N_]; mx = fmaxf(mx, l[r]); }
    float s = 0.f;
#pragma unroll
    for (int r = 0; r < K_; ++r) { l[r] = expf(l[r] - mx); s += l[r]; }
    const float inv = 1.f / s;
    float s2 = 0.f;
#pragma unroll
    for (int r = 0; r < K_; ++r) { l[r] = l[r] * inv + EPS_; s2 += l[r]; }
    const float inv2 = 1.f / s2;
#pragma unroll
    for (int r = 0; r < K_; ++r) {
        const float a = l[r] * inv2;
        attn_out[((size_t)b * K_ + r) * N_ + tok] = a;
        attn_pad[((size_t)b * 16 + r) * N_ + tok] = (_Float16)a;
    }
}

// ---------------------------------------------------------------------------
// Small elementwise kernels
// ---------------------------------------------------------------------------
__global__ void cvt_f32_f16(const float* __restrict__ s, _Float16* __restrict__ d, int n)
{ int i = blockIdx.x * blockDim.x + threadIdx.x; if (i < n) d[i] = (_Float16)s[i]; }

__global__ void zero_f16(_Float16* __restrict__ p, int n)
{ int i = blockIdx.x * blockDim.x + threadIdx.x; if (i < n) p[i] = (_Float16)0.f; }

__global__ void init_slots(const float* __restrict__ mu, const float* __restrict__ lsig,
                           const float* __restrict__ noise,
                           float* __restrict__ slots, _Float16* __restrict__ slots16)
{
    int i = blockIdx.x * blockDim.x + threadIdx.x;
    if (i >= BK_ * D_) return;
    int d = i & (D_ - 1);
    float s = mu[d] + expf(lsig[d]) * noise[i];
    slots[i] = s; slots16[i] = (_Float16)s;
}

__global__ void gru_combine(const float* __restrict__ gi, const float* __restrict__ gh,
                            const float* __restrict__ slots, float* __restrict__ h)
{
    int i = blockIdx.x * blockDim.x + threadIdx.x;
    if (i >= BK_ * D_) return;
    int row = i >> 8, d = i & 255;
    size_t base = (size_t)row * (3 * D_);
    float ir = gi[base + d], iz = gi[base + D_ + d], inn = gi[base + 2 * D_ + d];
    float hr = gh[base + d], hz = gh[base + D_ + d], hn  = gh[base + 2 * D_ + d];
    float r = 1.f / (1.f + expf(-(ir + hr)));
    float z = 1.f / (1.f + expf(-(iz + hz)));
    float n = tanhf(inn + r * hn);
    h[i] = (1.f - z) * n + z * slots[i];
}

__global__ void add_final(const float* __restrict__ h, const float* __restrict__ m2,
                          float* __restrict__ slots, _Float16* __restrict__ slots16,
                          float* __restrict__ out)
{
    int i = blockIdx.x * blockDim.x + threadIdx.x;
    if (i >= BK_ * D_) return;
    float s = h[i] + m2[i];
    slots[i] = s; slots16[i] = (_Float16)s; out[i] = s;
}

// ---------------------------------------------------------------------------
extern "C" void kernel_launch(void* const* d_in, const int* in_sizes, int n_in,
                              void* d_out, int out_size, void* d_ws, size_t ws_size,
                              hipStream_t stream)
{
    const float* x      = (const float*)d_in[0];
    const float* noise  = (const float*)d_in[1];
    const float* nx_g   = (const float*)d_in[2];
    const float* nx_b   = (const float*)d_in[3];
    const float* ns_g   = (const float*)d_in[4];
    const float* ns_b   = (const float*)d_in[5];
    const float* mu     = (const float*)d_in[6];
    const float* lsig   = (const float*)d_in[7];
    const float* Wq     = (const float*)d_in[8];
    const float* Wk     = (const float*)d_in[9];
    const float* Wv     = (const float*)d_in[10];
    const float* W_ih   = (const float*)d_in[11];
    const float* W_hh   = (const float*)d_in[12];
    const float* b_ih   = (const float*)d_in[13];
    const float* b_hh   = (const float*)d_in[14];
    const float* mlp_g  = (const float*)d_in[15];
    const float* mlp_b  = (const float*)d_in[16];
    const float* W1     = (const float*)d_in[17];
    const float* b1     = (const float*)d_in[18];
    const float* W2     = (const float*)d_in[19];
    const float* b2     = (const float*)d_in[20];

    float* out_slots = (float*)d_out;
    float* out_attn  = out_slots + BK_ * D_;

    // ---- workspace carve-up (256-B aligned) ----
    char* w = (char*)d_ws; size_t o = 0;
    auto alloc = [&](size_t bytes) -> char* {
        char* p = w + o; o += (bytes + 255) & ~(size_t)255; return p;
    };
    _Float16* wq16   = (_Float16*)alloc(D_ * D_ * 2);
    _Float16* wk16   = (_Float16*)alloc(D_ * D_ * 2);
    _Float16* wv16   = (_Float16*)alloc(D_ * D_ * 2);
    _Float16* wih16  = (_Float16*)alloc(3 * D_ * D_ * 2);
    _Float16* whh16  = (_Float16*)alloc(3 * D_ * D_ * 2);
    _Float16* w1_16  = (_Float16*)alloc(H_ * D_ * 2);
    _Float16* w2_16  = (_Float16*)alloc(D_ * H_ * 2);
    float*    slots  = (float*)   alloc(BK_ * D_ * 4);
    _Float16* slots16= (_Float16*)alloc(BK_ * D_ * 2);
    _Float16* s16    = (_Float16*)alloc(BK_ * D_ * 2);
    _Float16* qpad16 = (_Float16*)alloc(B_ * 16 * D_ * 2);
    _Float16* attnp  = (_Float16*)alloc((size_t)B_ * 16 * N_ * 2);
    _Float16* upd16  = (_Float16*)alloc(BK_ * D_ * 2);
    float*    gi     = (float*)   alloc(BK_ * 3 * D_ * 4);
    float*    gh     = (float*)   alloc(BK_ * 3 * D_ * 4);
    float*    hbuf   = (float*)   alloc(BK_ * D_ * 4);
    _Float16* mh16   = (_Float16*)alloc(BK_ * D_ * 2);
    _Float16* t16    = (_Float16*)alloc(BK_ * H_ * 2);
    float*    m2     = (float*)   alloc(BK_ * D_ * 4);
    _Float16* k16    = (_Float16*)alloc((size_t)B_ * N_ * D_ * 2);
    _Float16* vt16   = (_Float16*)alloc((size_t)B_ * D_ * N_ * 2);
    // xn16 (dead after projections) shares space with per-iter logits
    char*     unionr = alloc((size_t)B_ * N_ * D_ * 2);
    _Float16* xn16   = (_Float16*)unionr;
    float*    logits = (float*)unionr;          // B*16*N fp32 = 8 MB < 64 MB

    auto cdiv = [](int a, int b) { return (a + b - 1) / b; };

    // ---- weights -> f16 ----
    cvt_f32_f16<<<cdiv(D_*D_,256),256,0,stream>>>(Wq,   wq16,  D_*D_);
    cvt_f32_f16<<<cdiv(D_*D_,256),256,0,stream>>>(Wk,   wk16,  D_*D_);
    cvt_f32_f16<<<cdiv(D_*D_,256),256,0,stream>>>(Wv,   wv16,  D_*D_);
    cvt_f32_f16<<<cdiv(3*D_*D_,256),256,0,stream>>>(W_ih, wih16, 3*D_*D_);
    cvt_f32_f16<<<cdiv(3*D_*D_,256),256,0,stream>>>(W_hh, whh16, 3*D_*D_);
    cvt_f32_f16<<<cdiv(H_*D_,256),256,0,stream>>>(W1,   w1_16, H_*D_);
    cvt_f32_f16<<<cdiv(D_*H_,256),256,0,stream>>>(W2,   w2_16, D_*H_);

    // padded buffers: zero so rows 8..15 of each batch stay 0 forever
    zero_f16<<<cdiv(B_*16*D_,256),256,0,stream>>>(qpad16, B_*16*D_);
    zero_f16<<<cdiv(B_*16*N_,256),256,0,stream>>>(attnp,  B_*16*N_);

    init_slots<<<cdiv(BK_*D_,256),256,0,stream>>>(mu, lsig, noise, slots, slots16);

    // ---- xn = LN(x) -> f16 ----
    ln256<<<B_*N_, 256, 0, stream>>>(x, nx_g, nx_b, xn16);

    // ---- k = xn @ Wk^T (f16) ; vT = (xn @ Wv^T)^T (f16) ----
    // M = 131072, N = 256 -> NF=16 so A streams from HBM once per GEMM.
    wmma_gemm<16,0,true,0><<<dim3(B_*N_/16,1,1),32,0,stream>>>(
        xn16, D_, 0, wk16, D_, 0, k16, D_, 0, D_, nullptr, 1.f);
    wmma_gemm<16,3,true,0><<<dim3(B_*N_/16,1,1),32,0,stream>>>(
        xn16, D_, 0, wv16, D_, 0, vt16, 0, 0, D_, nullptr, 1.f);

    for (int it = 0; it < 3; ++it) {
        // s = LN(slots)
        ln256<<<BK_, 256, 0, stream>>>(slots, ns_g, ns_b, s16);
        // q = s @ Wq^T -> padded [B,16,D] f16 (rows 8..15 stay zero)
        wmma_gemm<4,1,true,0><<<dim3(BK_/16, D_/64, 1),32,0,stream>>>(
            s16, D_, 0, wq16, D_, 0, qpad16, D_, 0, D_, nullptr, 1.f);
        // logits[b] = q_pad[b] @ k[b]^T * SCALE  (batched, M=16, N=4096)
        wmma_gemm<4,0,false,0><<<dim3(1, N_/64, B_),32,0,stream>>>(
            qpad16, D_, (size_t)16*D_, k16, D_, (size_t)N_*D_,
            logits, N_, (size_t)16*N_, D_, nullptr, SCALE_);
        // softmax over slots (+EPS, renorm); attn -> d_out (fp32) + padded f16
        softmax_slots<<<cdiv(B_*N_,256),256,0,stream>>>(logits, out_attn, attnp);
        // updates[b] = attn[b] @ v[b]  via Bw = vT[b]   (compact rows 0..7)
        wmma_gemm<4,2,true,0><<<dim3(1, D_/64, B_),32,0,stream>>>(
            attnp, N_, (size_t)16*N_, vt16, N_, (size_t)D_*N_,
            upd16, D_, (size_t)K_*D_, N_, nullptr, 1.f);
        // GRU gates
        wmma_gemm<4,0,false,0><<<dim3(BK_/16, 3*D_/64, 1),32,0,stream>>>(
            upd16, D_, 0, wih16, D_, 0, gi, 3*D_, 0, D_, b_ih, 1.f);
        wmma_gemm<4,0,false,0><<<dim3(BK_/16, 3*D_/64, 1),32,0,stream>>>(
            slots16, D_, 0, whh16, D_, 0, gh, 3*D_, 0, D_, b_hh, 1.f);
        gru_combine<<<cdiv(BK_*D_,256),256,0,stream>>>(gi, gh, slots, hbuf);
        // MLP: m2 = GELU(LN(h) @ W1^T + b1) @ W2^T + b2 ; slots = h + m2
        ln256<<<BK_, 256, 0, stream>>>(hbuf, mlp_g, mlp_b, mh16);
        wmma_gemm<4,0,true,1><<<dim3(BK_/16, H_/64, 1),32,0,stream>>>(
            mh16, D_, 0, w1_16, D_, 0, t16, H_, 0, D_, b1, 1.f);
        wmma_gemm<4,0,false,0><<<dim3(BK_/16, D_/64, 1),32,0,stream>>>(
            t16, H_, 0, w2_16, H_, 0, m2, D_, 0, H_, b2, 1.f);
        add_final<<<cdiv(BK_*D_,256),256,0,stream>>>(hbuf, m2, slots, slots16, out_slots);
    }
}
